// BahdanauAttention_40699110097674
// MI455X (gfx1250) — compile-verified, matrix operand order checked by reading, not run on hardware
//
#include <hip/hip_runtime.h>
#include <hip/hip_bf16.h>
#include <math.h>

// Problem dims (fixed by setup_inputs)
#define BATCH 64
#define FDIM  4096
#define EDIM  256
#define HDIM  512
#define UDIM  512

typedef __attribute__((ext_vector_type(16))) __bf16 v16bf;
typedef __attribute__((ext_vector_type(8)))  float  v8f;

__device__ __forceinline__ unsigned short f2bf(float f) {
    unsigned int u = __builtin_bit_cast(unsigned int, f);
    u += 0x7FFFu + ((u >> 16) & 1u);   // round-to-nearest-even
    return (unsigned short)(u >> 16);
}

// ---------------------------------------------------------------------------
// Kernel 1: convert W1 (E x U fp32, row-major) into bf16 tiles laid out in the
// per-lane WMMA-B order. Block index = kt*(U/16)+nt ; within a block lane L
// owns column n = nt*16 + (L&15), K-chunks {kb..kb+7} and {kb+16..kb+23},
// kb = kt*32 + (L>>4)*8, stored as 16 contiguous bf16 per lane.
// ---------------------------------------------------------------------------
__global__ void __launch_bounds__(256) prep_w1(const float* __restrict__ W1,
                                               unsigned short* __restrict__ w1t) {
    int idx = blockIdx.x * 256 + threadIdx.x;        // 131072 elements
    int i  = idx & 15;
    int L  = (idx >> 4) & 31;
    int nt = (idx >> 9) & 31;                        // U/16 = 32
    int kt = idx >> 14;                              // E/32 = 8
    int n  = nt * 16 + (L & 15);
    int kb = kt * 32 + (L >> 4) * 8;
    int k  = kb + (i < 8 ? i : 8 + i);               // {0..7} then {16..23}
    w1t[idx] = f2bf(W1[k * UDIM + n]);
}

// ---------------------------------------------------------------------------
// Kernel 2: proj_h[b,u] = hidden[b,:] @ W2[:,u] + W2_b[u]   (64x512, tiny)
// ---------------------------------------------------------------------------
__global__ void __launch_bounds__(256) proj_h_kernel(const float* __restrict__ hidden,
                                                     const float* __restrict__ W2,
                                                     const float* __restrict__ W2b,
                                                     float* __restrict__ ph) {
    int idx = blockIdx.x * 256 + threadIdx.x;        // 32768
    int b = idx >> 9;
    int u = idx & (UDIM - 1);
    const float* hr = hidden + (size_t)b * HDIM;
    float acc = W2b[u];
#pragma unroll 4
    for (int k = 0; k < HDIM; ++k)
        acc += hr[k] * W2[(size_t)k * UDIM + u];
    ph[idx] = acc;
}

// ---------------------------------------------------------------------------
// Kernel 3: main fused GEMM + tanh + V-dot.
// One block = one (b, 64-row) super-tile: 4 M-tiles of 16 rows.
// 8 waves; each wave owns 4 N-tiles (64 of the U=512 columns) and reuses its
// B register fragments across the 4 M-tiles each K-step (16 wmma : 8 vmem : 8 ds).
// ---------------------------------------------------------------------------
__global__ void __launch_bounds__(256) attn_main(const float* __restrict__ feat,
                                                 const float* __restrict__ W1b,
                                                 const float* __restrict__ Vw,
                                                 const float* __restrict__ Vb,
                                                 const unsigned short* __restrict__ w1t,
                                                 const float* __restrict__ ph,
                                                 float* __restrict__ scores) {
    __shared__ __align__(32) unsigned short la[4 * 8 * 32 * 16];   // 4 A tiles, 32 KB
    __shared__ float ssc[64];

    const int tid   = threadIdx.x;
    const int b     = blockIdx.x >> 6;                 // F/64 = 64 super-tiles / batch
    const int st    = blockIdx.x & 63;
    const int frow0 = st * 64;

    if (tid < 64) ssc[tid] = 0.0f;

    // ---- stage A (64 rows x 256 K, fp32 -> bf16) into WMMA-A per-lane layout
    {
        const int L  = tid & 31;
        const int kt = tid >> 5;                       // 0..7
        const int kb = kt * 32 + (L >> 4) * 8;
#pragma unroll
        for (int mt = 0; mt < 4; ++mt) {
            const int row = frow0 + mt * 16 + (L & 15);
            const float* src = feat + ((size_t)b * FDIM + row) * EDIM + kb;
            float4 c0 = *(const float4*)(src);
            float4 c1 = *(const float4*)(src + 4);
            float4 c2 = *(const float4*)(src + 16);
            float4 c3 = *(const float4*)(src + 20);
            unsigned short* dst = &la[mt * 4096 + tid * 16];
            dst[0]  = f2bf(c0.x); dst[1]  = f2bf(c0.y); dst[2]  = f2bf(c0.z); dst[3]  = f2bf(c0.w);
            dst[4]  = f2bf(c1.x); dst[5]  = f2bf(c1.y); dst[6]  = f2bf(c1.z); dst[7]  = f2bf(c1.w);
            dst[8]  = f2bf(c2.x); dst[9]  = f2bf(c2.y); dst[10] = f2bf(c2.z); dst[11] = f2bf(c2.w);
            dst[12] = f2bf(c3.x); dst[13] = f2bf(c3.y); dst[14] = f2bf(c3.z); dst[15] = f2bf(c3.w);
        }
    }
    __syncthreads();

    const int lane = tid & 31;
    const int w    = tid >> 5;                         // wave id 0..7

    v8f acc[4][4] = {};                                // [mtile][ntile], 128 VGPRs

#pragma unroll
    for (int kt = 0; kt < 8; ++kt) {
        // load this wave's 4 B fragments once per K-step
        v16bf bm[4];
#pragma unroll
        for (int t = 0; t < 4; ++t) {
            const int nt = w * 4 + t;
            bm[t] = *(const v16bf*)(w1t + ((size_t)(kt * 32 + nt) * 32 + lane) * 16);
        }
        // reuse them across 4 M-tiles
#pragma unroll
        for (int mt = 0; mt < 4; ++mt) {
            v16bf a = *(const v16bf*)&la[mt * 4096 + (kt * 32 + lane) * 16];
#pragma unroll
            for (int t = 0; t < 4; ++t)
                acc[mt][t] = __builtin_amdgcn_wmma_f32_16x16x32_bf16(
                    false, a, false, bm[t], (short)0, acc[mt][t], false, false);
        }
    }

    // ---- epilogue: +bias +proj_h, tanh, * V_w, reduce over U
    const int half = lane >> 4;        // C layout: VGPR j = row (j + 8*half)
    const int nlo  = lane & 15;

#pragma unroll
    for (int mt = 0; mt < 4; ++mt) {
        float partial[8];
#pragma unroll
        for (int j = 0; j < 8; ++j) partial[j] = 0.0f;
#pragma unroll
        for (int t = 0; t < 4; ++t) {
            const int u = (w * 4 + t) * 16 + nlo;
            const float bias = W1b[u] + ph[(size_t)b * UDIM + u];
            const float vw   = Vw[u];
#pragma unroll
            for (int j = 0; j < 8; ++j)
                partial[j] += tanhf(acc[mt][t][j] + bias) * vw;
        }
#pragma unroll
        for (int j = 0; j < 8; ++j) {
            float p = partial[j];
            p += __shfl_xor(p, 1);
            p += __shfl_xor(p, 2);
            p += __shfl_xor(p, 4);
            p += __shfl_xor(p, 8);     // masks <16: stays within each 16-lane half
            if (nlo == 0) atomicAdd(&ssc[mt * 16 + j + half * 8], p);
        }
    }
    __syncthreads();

    if (tid < 64)
        scores[(size_t)b * FDIM + frow0 + tid] = ssc[tid] + Vb[0];
}

// ---------------------------------------------------------------------------
// Kernel 4: softmax over F per batch; writes attention weights into d_out.
// ---------------------------------------------------------------------------
__global__ void __launch_bounds__(256) softmax_k(const float* __restrict__ scores,
                                                 float* __restrict__ wout) {
    __shared__ float red[256];
    const int b = blockIdx.x;
    const int t = threadIdx.x;
    const float* s = scores + (size_t)b * FDIM;
    float* wb = wout + (size_t)b * FDIM;

    float m = -INFINITY;
    for (int f = t; f < FDIM; f += 256) m = fmaxf(m, s[f]);
    red[t] = m; __syncthreads();
    for (int o = 128; o > 0; o >>= 1) {
        if (t < o) red[t] = fmaxf(red[t], red[t + o]);
        __syncthreads();
    }
    const float mx = red[0];
    __syncthreads();

    float sum = 0.0f;
    for (int f = t; f < FDIM; f += 256) {
        float e = expf(s[f] - mx);
        wb[f] = e;
        sum += e;
    }
    red[t] = sum; __syncthreads();
    for (int o = 128; o > 0; o >>= 1) {
        if (t < o) red[t] += red[t + o];
        __syncthreads();
    }
    const float inv = 1.0f / red[0];
    for (int f = t; f < FDIM; f += 256) wb[f] *= inv;
}

// ---------------------------------------------------------------------------
// Kernel 5: context[b,e] = sum_f w[b,f] * features[b,f,e]
// ---------------------------------------------------------------------------
__global__ void __launch_bounds__(256) zero_ctx(float* __restrict__ ctx) {
    int i = blockIdx.x * 256 + threadIdx.x;
    if (i < BATCH * EDIM) ctx[i] = 0.0f;
}

__global__ void __launch_bounds__(256) context_k(const float* __restrict__ feat,
                                                 const float* __restrict__ wout,
                                                 float* __restrict__ ctx) {
    const int b  = blockIdx.x >> 4;
    const int fc = blockIdx.x & 15;                   // 16 f-chunks of 256
    const int e  = threadIdx.x;                       // E == 256
    const float* w  = wout + (size_t)b * FDIM + fc * 256;
    const float* fp = feat + ((size_t)b * FDIM + fc * 256) * EDIM + e;
    float acc = 0.0f;
#pragma unroll 4
    for (int f = 0; f < 256; ++f)
        acc += w[f] * fp[(size_t)f * EDIM];
    atomicAdd(&ctx[b * EDIM + e], acc);
}

// ---------------------------------------------------------------------------
extern "C" void kernel_launch(void* const* d_in, const int* in_sizes, int n_in,
                              void* d_out, int out_size, void* d_ws, size_t ws_size,
                              hipStream_t stream) {
    (void)in_sizes; (void)n_in; (void)out_size; (void)ws_size;

    const float* feat   = (const float*)d_in[0];   // (B,F,E)
    const float* hidden = (const float*)d_in[1];   // (B,H)
    const float* W1_w   = (const float*)d_in[2];   // (E,U)
    const float* W1_b   = (const float*)d_in[3];   // (U,)
    const float* W2_w   = (const float*)d_in[4];   // (H,U)
    const float* W2_b   = (const float*)d_in[5];   // (U,)
    const float* V_w    = (const float*)d_in[6];   // (U,1)
    const float* V_b    = (const float*)d_in[7];   // (1,)

    float* ctx  = (float*)d_out;                        // (B,E)
    float* wout = ctx + (size_t)BATCH * EDIM;           // (B,F,1)

    // workspace layout
    char* ws = (char*)d_ws;
    unsigned short* w1t = (unsigned short*)ws;                       // 256 KB
    float* ph     = (float*)(ws + 262144);                           // 128 KB
    float* scores = (float*)(ws + 262144 + 131072);                  // 1 MB

    prep_w1<<<512, 256, 0, stream>>>(W1_w, w1t);
    proj_h_kernel<<<(BATCH * UDIM) / 256, 256, 0, stream>>>(hidden, W2_w, W2_b, ph);
    zero_ctx<<<(BATCH * EDIM + 255) / 256, 256, 0, stream>>>(ctx);

    attn_main<<<BATCH * (FDIM / 64), 256, 0, stream>>>(feat, W1_b, V_w, V_b,
                                                       w1t, ph, scores);   // 4096 blocks
    softmax_k<<<BATCH, 256, 0, stream>>>(scores, wout);
    context_k<<<BATCH * 16, 256, 0, stream>>>(feat, wout, ctx);
}
